// HybridClassifier_65481071409312
// MI455X (gfx1250) — matrix operand order repaired and checked by reading, not compiled, not run on hardware
//
#include <hip/hip_runtime.h>

// ---------------------------------------------------------------------------
// Hybrid quantum-classical classifier for MI455X (gfx1250, wave32).
//   Kernel 1: 10-qubit state-vector simulation, 1 block (1024 thr) per sample,
//             state kept in LDS, deterministic tree reductions -> qfeat (B x 10)
//   Kernel 2: weights DMA'd via TDM (tensor_load_to_lds + s_wait_tensorcnt),
//             5-layer MLP via v_wmma_f32_16x16x32_f16 with pre-swizzled
//             per-lane LDS fragments (ds_load_b128 operand feeds),
//             fused final 20->2 projection. One wave (32 thr) per 16-row tile.
// ---------------------------------------------------------------------------

typedef __attribute__((ext_vector_type(16))) _Float16 v16h;
typedef __attribute__((ext_vector_type(8)))  float    v8f;
typedef __attribute__((ext_vector_type(4)))  unsigned u32x4;
typedef __attribute__((ext_vector_type(8)))  int      i32x8;
typedef __attribute__((ext_vector_type(4)))  int      i32x4;

#define NQ     10
#define QDEPTH 4
#define NSTATE 1024   // 2^NQ

// ------------------------- quantum feature kernel --------------------------
__global__ __launch_bounds__(NSTATE)
void quantum_kernel(const float* __restrict__ x, const float* __restrict__ theta,
                    float* __restrict__ qfeat, int B) {
    __shared__ float re[NSTATE];
    __shared__ float im[NSTATE];
    __shared__ float cs[NQ], sn[NQ];                   // RX(x) half-angle cos/sin
    __shared__ float gc[QDEPTH * NQ], gs[QDEPTH * NQ]; // RY(theta) half-angle
    __shared__ float sums[NQ + 1];

    const int b = blockIdx.x;
    const int i = threadIdx.x;     // amplitude index, 0..1023

    if (i < NQ) {
        float h = 0.5f * x[b * NQ + i];
        cs[i] = cosf(h);
        sn[i] = sinf(h);
    }
    if (i < QDEPTH * NQ) {
        float h = 0.5f * theta[i];
        gc[i] = cosf(h);
        gs[i] = sinf(h);
    }
    __syncthreads();

    // Initial product state: amp(i) = prod_q (bit? -i*sin : cos)
    //                              = magnitude * (-i)^popcount(set bits)
    float m = 1.0f;
    int   k = 0;
#pragma unroll
    for (int q = 0; q < NQ; ++q) {
        if ((i >> q) & 1) { m *= sn[q]; ++k; }
        else              { m *= cs[q]; }
    }
    k &= 3;
    re[i] = (k == 0) ? m : ((k == 2) ? -m : 0.0f);
    im[i] = (k == 3) ? m : ((k == 1) ? -m : 0.0f);

    // Diagonal of prod_q CZ(q,q+1): sign = (-1)^(# adjacent 11 pairs)
    const int   adj = i & (i >> 1) & 0x1FF;
    const float czs = (__popc(adj) & 1) ? -1.0f : 1.0f;
    __syncthreads();

    for (int l = 0; l < QDEPTH; ++l) {
        for (int q = 0; q < NQ; ++q) {
            const float ct = gc[l * NQ + q];
            const float st = gs[l * NQ + q];
            const int   j  = i ^ (1 << q);   // paired amplitude
            float ar = re[i], ai = im[i];
            float br = re[j], bi = im[j];
            __syncthreads();
            // RY: new(bit0) = ct*a0 - st*a1 ; new(bit1) = st*a0 + ct*a1
            const float s = ((i >> q) & 1) ? st : -st;
            re[i] = ct * ar + s * br;
            im[i] = ct * ai + s * bi;
            __syncthreads();
        }
        re[i] *= czs;
        im[i] *= czs;
        __syncthreads();
    }

    const float p = re[i] * re[i] + im[i] * im[i];
    __syncthreads();

    // 11 deterministic tree reductions: total prob + per-qubit P(bit=1)
    for (int t = 0; t <= NQ; ++t) {
        re[i] = (t == NQ || ((i >> t) & 1)) ? p : 0.0f;
        __syncthreads();
        for (int s = NSTATE / 2; s > 0; s >>= 1) {
            if (i < s) re[i] += re[i + s];
            __syncthreads();
        }
        if (i == 0) sums[t] = re[0];
        __syncthreads();
    }

    // column j of z_matrix corresponds to qubit (NQ-1-j): <Z> = total - 2*P(1)
    if (i < NQ) qfeat[b * NQ + i] = sums[NQ] - 2.0f * sums[NQ - 1 - i];
}

// --------------------------- WMMA MLP kernel -------------------------------
__global__ __launch_bounds__(32)
void mlp_kernel(const float* __restrict__ x,  const float* __restrict__ Ws,
                const float* __restrict__ bs, const float* __restrict__ Wf,
                const float* __restrict__ bf, const float* __restrict__ qfeat,
                float* __restrict__ out, int B) {
    __shared__ __attribute__((aligned(16))) float    WsRaw[512];               // TDM landing (500 used)
    __shared__ __attribute__((aligned(16))) _Float16 Bfrag[QDEPTH + 1][32][16]; // per-lane B operands
    __shared__ __attribute__((aligned(16))) _Float16 Afrag[32][16];             // per-lane A operand
    __shared__ float bias[QDEPTH + 1][16];
    __shared__ float Dtile[16][16];            // D relayout staging (M x N)

    const int lane = threadIdx.x;              // 0..31, one full wave
    const int row0 = blockIdx.x * 16;

    __builtin_prefetch(x + row0 * NQ, 0, 0);   // global_prefetch_b8

    // ---- TDM: DMA the 500 contiguous fp32 weights into LDS (D# per ISA 8.x) ----
    {
        const unsigned long long ga  = (unsigned long long)(uintptr_t)Ws;
        const unsigned           lds = (unsigned)(uintptr_t)&WsRaw[0];
        u32x4 g0 = { 1u,                                           // count=1, no gather
                     lds,                                          // lds_addr [63:32]
                     (unsigned)(ga & 0xFFFFFFFFull),               // global_addr lo
                     (unsigned)((ga >> 32) & 0x01FFFFFFull)        // global_addr hi (bits 56:32)
                         | (2u << 30) };                           // type=2 ("image")
        i32x8 g1 = { (int)(2u << 16),      // workgroup_mask=0, data_size=2 (4 bytes)
                     (int)(500u << 16),    // tensor_dim0[15:0] at bits [63:48]
                     (int)(1u << 16),      // tensor_dim0 hi=0, tensor_dim1=1 (lo16)
                     (int)(500u << 16),    // tensor_dim1 hi=0, tile_dim0=500
                     1,                    // tile_dim1=1, tile_dim2=0
                     500,                  // tensor_dim0_stride lo32
                     0, 0 };               // stride hi / dim1_stride = 0
        i32x4 gz = { 0, 0, 0, 0 };
#if defined(__clang_major__) && (__clang_major__ >= 23)
        i32x8 gz8 = { 0, 0, 0, 0, 0, 0, 0, 0 };
        __builtin_amdgcn_tensor_load_to_lds(g0, g1, gz, gz, gz8, 0);
#else
        __builtin_amdgcn_tensor_load_to_lds(g0, g1, gz, gz, 0);
#endif
    }

    // ---- overlap with TDM: x tile -> A fragments (pre-swizzled per-lane) ----
    // A-fragment mapping (16-bit A 16x32, ISA 7.12.2): element (m,k) lives at
    //   lane = m + 16*((k>>3)&1),  j = (k&7) + 8*((k>>4)&1)
    // writer enumerates (lane, j) and inverts: k = 8*((lane>>4)) + (j&7) + 16*(j>>3)
    for (int t = lane; t < 32 * 16; t += 32) {
        const int fl = t >> 4, j = t & 15;
        const int m  = fl & 15;
        const int k  = 8 * (fl >> 4) + (j & 7) + 16 * (j >> 3);
        const int row = row0 + m;
        const float v = (k < NQ && row < B) ? x[row * NQ + k] : 0.0f;
        Afrag[fl][j] = (_Float16)v;
    }
    for (int t = lane; t < (QDEPTH + 1) * 16; t += 32) {
        const int l = t >> 4, n = t & 15;
        bias[l][n] = (n < NQ) ? bs[l * NQ + n] : 0.0f;
    }

    __builtin_amdgcn_s_wait_tensorcnt(0);
    __syncthreads();

    // ---- weights from LDS -> B fragments (pre-swizzled per-lane) ----
    // B 32x16 layout: element (k,n) at lane = n + 16*((k>>3)&1), j = (k&7) + 8*((k>>4)&1).
    // B[k][n] = W_l[n][k] so that D = A x B computes h @ W^T.
    for (int t = lane; t < (QDEPTH + 1) * 32 * 16; t += 32) {
        const int l = t >> 9, fl = (t >> 4) & 31, j = t & 15;
        const int n = fl & 15;
        const int k = 8 * (fl >> 4) + (j & 7) + 16 * (j >> 3);
        const float w = (k < NQ && n < NQ) ? WsRaw[(l * NQ + n) * NQ + k] : 0.0f;
        (&Bfrag[0][0][0])[t] = (_Float16)w;
    }
    __syncthreads();

    const int g  = lane >> 4;   // half-group
    const int ln = lane & 15;

    v16h a = *(const v16h*)&Afrag[lane][0];     // 2x ds_load_b128
    v8f  d;
    for (int l = 0; l <= QDEPTH; ++l) {
        v16h bfr = *(const v16h*)&Bfrag[l][lane][0];
        v8f  c;
#pragma unroll
        for (int r = 0; r < 8; ++r) c[r] = bias[l][ln];     // N=ln column bias
        d = __builtin_amdgcn_wmma_f32_16x16x32_f16(
                /*neg_a=*/false, a, /*neg_b=*/false, bfr,
                /*c_mod=*/(short)0, c, /*reuse_a=*/false, /*reuse_b=*/false);
        if (l < QDEPTH) {
            __syncthreads();
#pragma unroll
            for (int r = 0; r < 8; ++r)                 // D layout: M = r + 8g, N = ln
                Dtile[r + 8 * g][ln] = fmaxf(d[r], 0.0f);
            __syncthreads();
            // next A: element j<8 -> K = 8g + j (K in 0..15); j>=8 -> K>=16 == 0
#pragma unroll
            for (int j = 0; j < 8; ++j) a[j] = (_Float16)Dtile[ln][8 * g + j];
#pragma unroll
            for (int j = 8; j < 16; ++j) a[j] = (_Float16)0.0f;
        }
    }

    // cls_out tile -> LDS, then fused final Linear(20 -> 2): one dot per lane
    __syncthreads();
#pragma unroll
    for (int r = 0; r < 8; ++r) Dtile[r + 8 * g][ln] = d[r];
    __syncthreads();

    const int o    = g;            // output class 0/1
    const int grow = row0 + ln;    // global batch row
    float acc = bf[o];
#pragma unroll
    for (int k = 0; k < NQ; ++k) acc += Dtile[ln][k] * Wf[o * 2 * NQ + k];
    if (grow < B) {
#pragma unroll
        for (int k = 0; k < NQ; ++k) acc += qfeat[grow * NQ + k] * Wf[o * 2 * NQ + NQ + k];
        out[grow * 2 + o] = acc;
    }
}

// ------------------------------- launcher ----------------------------------
extern "C" void kernel_launch(void* const* d_in, const int* in_sizes, int n_in,
                              void* d_out, int out_size, void* d_ws, size_t ws_size,
                              hipStream_t stream) {
    const float* x  = (const float*)d_in[0];   // (B, 10)
    const float* Ws = (const float*)d_in[1];   // (5, 10, 10)
    const float* bs = (const float*)d_in[2];   // (5, 10)
    const float* th = (const float*)d_in[3];   // (4, 10)
    const float* Wf = (const float*)d_in[4];   // (2, 20)
    const float* bf = (const float*)d_in[5];   // (2,)
    float* out = (float*)d_out;                // (B, 2)

    const int B = in_sizes[0] / NQ;
    float* qfeat = (float*)d_ws;               // (B, 10) staging

    quantum_kernel<<<B, NSTATE, 0, stream>>>(x, th, qfeat, B);

    const int tiles = (B + 15) / 16;
    mlp_kernel<<<tiles, 32, 0, stream>>>(x, Ws, bs, Wf, bf, qfeat, out, B);
}
